// Environment_16509854286015
// MI455X (gfx1250) — matrix operand order repaired and checked by reading, not compiled, not run on hardware
//
#include <hip/hip_runtime.h>
#include <hip/hip_bf16.h>

// ---------------------------------------------------------------------------
// Batched Ising energy  E_b = s_b^T J s_b,  s = 2*state - 1 in {-1,+1}
// B = 1024 configurations, N = 2048 spins.
//
// T = S_bf16 @ J_bf16 via v_wmma_f32_16x16x32_bf16 (f32 accum), 2x2 register
// blocking (32x32 output per wave, 4 accumulators) to halve L2 fragment
// traffic per WMMA; fused epilogue E_b += sum_j T[b,j] * s[b,j].
// ---------------------------------------------------------------------------

typedef __bf16 bf16_t;
typedef __attribute__((ext_vector_type(16))) __bf16 v16bf;
typedef __attribute__((ext_vector_type(8)))  __bf16 v8bf;
typedef __attribute__((ext_vector_type(8)))  float  v8f;

static constexpr int kB = 1024;   // batch of spin configurations
static constexpr int kN = 2048;   // number of spins

// ---------------------------------------------------------------------------
// state {0,1} f32  ->  spins {-1,+1} bf16 (exact in bf16)
// ---------------------------------------------------------------------------
__global__ void spins_to_bf16_kernel(const float* __restrict__ state,
                                     bf16_t* __restrict__ sbf, int n) {
  int i = blockIdx.x * blockDim.x + threadIdx.x;
  if (i < n) sbf[i] = (bf16_t)(2.0f * state[i] - 1.0f);
}

// ---------------------------------------------------------------------------
// J (f32, row major)  ->  Jt (bf16, transposed), LDS-tiled 32x32 transpose.
// Jt[n][k] = J[k][n] so B-fragment loads (fixed column n, running k) are
// contiguous 32-byte reads.
// ---------------------------------------------------------------------------
__global__ void transpose_to_bf16_kernel(const float* __restrict__ J,
                                         bf16_t* __restrict__ Jt) {
  __shared__ bf16_t tile[32][33];           // +1 pad: avoid bank conflicts
  const int tx = threadIdx.x;               // 0..31
  const int ty = threadIdx.y;               // 0..7
  const int x0 = blockIdx.x * 32;           // column block in J
  const int y0 = blockIdx.y * 32;           // row block in J
#pragma unroll
  for (int i = 0; i < 32; i += 8)
    tile[ty + i][tx] = (bf16_t)J[(size_t)(y0 + ty + i) * kN + (x0 + tx)];
  __syncthreads();
#pragma unroll
  for (int i = 0; i < 32; i += 8)
    Jt[(size_t)(x0 + ty + i) * kN + (y0 + tx)] = tile[tx][ty + i];
}

// ---------------------------------------------------------------------------
// Zero-init the output (harness poisons d_out with 0xAA).
// ---------------------------------------------------------------------------
__global__ void zero_out_kernel(float* __restrict__ out, int n) {
  int i = blockIdx.x * blockDim.x + threadIdx.x;
  if (i < n) out[i] = 0.0f;
}

// ---------------------------------------------------------------------------
// Helper: load one 16x32 bf16 A/B fragment worth of K from ptr (K-contiguous
// row) in this file's contiguous k order (k = 16*half + s), shared by A and B.
//   A 16x32: lane L -> M = L&15;  lanes 0-15: K = 0..15,
//                                 lanes 16-31: K = 16..31
//     -> two 16B loads at +0 and +8 elements from a (half?16:0) base.
//   B 32x16: lane L -> N = L&15;  lanes 0-15: K = 0..15,
//                                 lanes 16-31: K = 16..31
//     -> one contiguous 32B load from a (half?16:0) base.
// ---------------------------------------------------------------------------
__device__ __forceinline__ v16bf load_a_frag(const bf16_t* p) {
  v8bf lo = *(const v8bf*)(p);
  v8bf hi = *(const v8bf*)(p + 8);
  return __builtin_shufflevector(lo, hi, 0, 1, 2, 3, 4, 5, 6, 7,
                                 8, 9, 10, 11, 12, 13, 14, 15);
}

__device__ __forceinline__ v8f wmma_bf16(v16bf a, v16bf b, v8f c) {
  return __builtin_amdgcn_wmma_f32_16x16x32_bf16(
      /*neg_a=*/false, a, /*neg_b=*/false, b,
      /*c_mod=*/(short)0, c, /*reuse_a=*/false, /*reuse_b=*/false);
}

// ---------------------------------------------------------------------------
// One wave per 32x32 tile of T = S @ J (2x2 blocking of 16x16 WMMA tiles).
// K-loop: 64 steps x 4 WMMAs; fused epilogue multiplies by s[b,j],
// shuffle-reduces rows, atomicAdd into E_b.
//   C/D f32 layout: VGPR r -> row r + 8*(L>=16), col = L&15
// ---------------------------------------------------------------------------
__global__ void __launch_bounds__(32)
ising_energy_wmma_kernel(const bf16_t* __restrict__ Sbf,
                         const bf16_t* __restrict__ Jt,
                         const float*  __restrict__ state,
                         float* __restrict__ out) {
  const int lane = threadIdx.x;         // 0..31, wave32
  const int half = lane >> 4;           // 0 or 1
  const int l15  = lane & 15;
  const int i0 = blockIdx.y * 32;       // batch-row tile origin (2 x 16)
  const int j0 = blockIdx.x * 32;       // spin-column tile origin (2 x 16)

  v8f acc[2][2] = {{{}, {}}, {{}, {}}};

  // A fragment bases: rows i0+l15 and i0+16+l15; half selects K sub-bank.
  const bf16_t* aBase0 = Sbf + (size_t)(i0 + l15) * kN + (half ? 16 : 0);
  const bf16_t* aBase1 = aBase0 + (size_t)16 * kN;
  // B fragment bases: Jt rows j0+l15 and j0+16+l15; half selects K half.
  const bf16_t* bBase0 = Jt + (size_t)(j0 + l15) * kN + (half ? 16 : 0);
  const bf16_t* bBase1 = bBase0 + (size_t)16 * kN;

  for (int k0 = 0; k0 < kN; k0 += 32) {
    v16bf a0 = load_a_frag(aBase0 + k0);
    v16bf a1 = load_a_frag(aBase1 + k0);
    v16bf b0 = *(const v16bf*)(bBase0 + k0);
    v16bf b1 = *(const v16bf*)(bBase1 + k0);
    acc[0][0] = wmma_bf16(a0, b0, acc[0][0]);
    acc[0][1] = wmma_bf16(a0, b1, acc[0][1]);
    acc[1][0] = wmma_bf16(a1, b0, acc[1][0]);
    acc[1][1] = wmma_bf16(a1, b1, acc[1][1]);
  }

  // Fused epilogue: E_b += sum_j T[b,j] * s[b,j].  The two column-block
  // accumulators that share a row are combined before the lane reduction,
  // so each wave issues only 16 atomics.
  const int col0 = j0 + l15;
  const int col1 = col0 + 16;
#pragma unroll
  for (int ia = 0; ia < 2; ++ia) {
    float partial[8];
#pragma unroll
    for (int r = 0; r < 8; ++r) {
      const int row = i0 + 16 * ia + r + (half ? 8 : 0);
      const float s0 = 2.0f * state[(size_t)row * kN + col0] - 1.0f;
      const float s1 = 2.0f * state[(size_t)row * kN + col1] - 1.0f;
      float v = acc[ia][0][r] * s0 + acc[ia][1][r] * s1;
      // reduce across the 16 lanes of this half-wave (same output row)
      v += __shfl_xor(v, 1, 32);
      v += __shfl_xor(v, 2, 32);
      v += __shfl_xor(v, 4, 32);
      v += __shfl_xor(v, 8, 32);
      partial[r] = v;
    }
    if (l15 == 0) {
#pragma unroll
      for (int r = 0; r < 8; ++r)
        atomicAdd(&out[i0 + 16 * ia + r + (half ? 8 : 0)], partial[r]);
    }
  }
}

// ---------------------------------------------------------------------------
// Launch: convert -> transpose -> zero -> WMMA GEMM + fused reduction.
// Workspace: Sbf (4 MB bf16) then Jt (8 MB bf16) = 12 MB total.
// ---------------------------------------------------------------------------
extern "C" void kernel_launch(void* const* d_in, const int* in_sizes, int n_in,
                              void* d_out, int out_size, void* d_ws, size_t ws_size,
                              hipStream_t stream) {
  (void)in_sizes; (void)n_in; (void)out_size; (void)ws_size;
  const float* state = (const float*)d_in[0];   // [B, N] f32 in {0,1}
  const float* J     = (const float*)d_in[1];   // [N, N] f32
  float* out = (float*)d_out;                   // [B] f32

  bf16_t* Sbf = (bf16_t*)d_ws;                               // B*N bf16
  bf16_t* Jt  = (bf16_t*)((char*)d_ws + (size_t)kB * kN * sizeof(bf16_t));

  {
    const int n = kB * kN;
    spins_to_bf16_kernel<<<(n + 255) / 256, 256, 0, stream>>>(state, Sbf, n);
  }
  {
    dim3 blk(32, 8), grd(kN / 32, kN / 32);
    transpose_to_bf16_kernel<<<grd, blk, 0, stream>>>(J, Jt);
  }
  zero_out_kernel<<<(kB + 255) / 256, 256, 0, stream>>>(out, kB);
  {
    dim3 grd(kN / 32, kB / 32);   // 64 x 32 tiles of 32x32, one wave each
    ising_energy_wmma_kernel<<<grd, 32, 0, stream>>>(Sbf, Jt, state, out);
  }
}
